// GarmentPersonCrossAttention_86895778332910
// MI455X (gfx1250) — compile-verified, matrix operand order checked by reading, not run on hardware
//
#include <hip/hip_runtime.h>

#define BATCH 4
#define SEQ_N 2048
#define SEQ_M 1024
#define DQv   1024
#define DCv   768
#define NHEAD 16
#define DHEAD 64
#define INNERv 1024
#define ATT_SCALE 0.125f

typedef _Float16 v16h __attribute__((ext_vector_type(16)));
typedef _Float16 v8h  __attribute__((ext_vector_type(8)));
typedef float    v8f  __attribute__((ext_vector_type(8)));

__device__ __forceinline__ v8f v8f_zero() {
  v8f z = {0.f,0.f,0.f,0.f,0.f,0.f,0.f,0.f};
  return z;
}

// Load one 16-bit WMMA A/B fragment for this lane.
// Per CDNA5 ISA 16-bit 16x32 layout: a lane holds K chunks {base..base+7} and
// {base+16..base+23}; caller passes p already offset by ks*32 + (laneHi?8:0).
__device__ __forceinline__ v16h ldfrag(const _Float16* p) {
  v8h lo = *(const v8h*)(p);
  v8h hi = *(const v8h*)(p + 16);
  return __builtin_shufflevector(lo, hi, 0,1,2,3,4,5,6,7,8,9,10,11,12,13,14,15);
}

__device__ __forceinline__ v8f wmma16x16x32(v16h a, v16h b, v8f c) {
  return __builtin_amdgcn_wmma_f32_16x16x32_f16(false, a, false, b, (short)0, c,
                                                false, false);
}

// ---------------------------------------------------------------------------
// Weight transpose + f32 -> f16 convert:  Wt[n][k] = (f16) W[k][n]
// ---------------------------------------------------------------------------
__global__ void wtrans_kernel(const float* __restrict__ W, _Float16* __restrict__ Wt,
                              int K, int N) {
  __shared__ float tile[32][33];
  int nt = blockIdx.x * 32, kt = blockIdx.y * 32;
  int tx = threadIdx.x, ty = threadIdx.y;
#pragma unroll
  for (int j = 0; j < 4; ++j)
    tile[ty + j * 8][tx] = W[(size_t)(kt + ty + j * 8) * N + nt + tx];
  __syncthreads();
#pragma unroll
  for (int j = 0; j < 4; ++j)
    Wt[(size_t)(nt + ty + j * 8) * K + kt + tx] = (_Float16)tile[tx][ty + j * 8];
}

// ---------------------------------------------------------------------------
// Per-batch transpose of V (f16):  Vt[b][c][m] = V[b*M + m][c]
// ---------------------------------------------------------------------------
__global__ void vtrans_kernel(const _Float16* __restrict__ V, _Float16* __restrict__ Vt) {
  __shared__ _Float16 tile[32][33];
  int b = blockIdx.z;
  int ct = blockIdx.x * 32, mt = blockIdx.y * 32;
  int tx = threadIdx.x, ty = threadIdx.y;
#pragma unroll
  for (int j = 0; j < 4; ++j)
    tile[ty + j * 8][tx] = V[((size_t)b * SEQ_M + mt + ty + j * 8) * INNERv + ct + tx];
  __syncthreads();
#pragma unroll
  for (int j = 0; j < 4; ++j)
    Vt[((size_t)b * INNERv + ct + ty + j * 8) * SEQ_M + mt + tx] = tile[tx][ty + j * 8];
}

// ---------------------------------------------------------------------------
// Row LayerNorm (f32 statistics) producing f16 rows.
// ---------------------------------------------------------------------------
template <int D>
__global__ void ln_f16_kernel(const float* __restrict__ x, const float* __restrict__ g,
                              const float* __restrict__ beta, _Float16* __restrict__ y) {
  __shared__ float s1[256], s2[256];
  int row = blockIdx.x, t = threadIdx.x;
  const float* xr = x + (size_t)row * D;
  float a = 0.f, b = 0.f;
  for (int i = t; i < D; i += 256) {
    float v = xr[i];
    a += v;
    b += v * v;
  }
  s1[t] = a;
  s2[t] = b;
  __syncthreads();
  for (int off = 128; off > 0; off >>= 1) {
    if (t < off) {
      s1[t] += s1[t + off];
      s2[t] += s2[t + off];
    }
    __syncthreads();
  }
  float mu = s1[0] * (1.0f / D);
  float var = s2[0] * (1.0f / D) - mu * mu;
  float rs = rsqrtf(var + 1e-5f);
  _Float16* yr = y + (size_t)row * D;
  for (int i = t; i < D; i += 256)
    yr[i] = (_Float16)((xr[i] - mu) * rs * g[i] + beta[i]);
}

// ---------------------------------------------------------------------------
// Residual f32 -> f16 into concat buffer cols [0, DQ)
// ---------------------------------------------------------------------------
__global__ void cvt_resid_kernel(const float* __restrict__ x, _Float16* __restrict__ y) {
  size_t i = (size_t)blockIdx.x * blockDim.x + threadIdx.x;
  if (i < (size_t)BATCH * SEQ_N * DQv) {
    size_t row = i >> 10, col = i & 1023;
    y[row * (2 * DQv) + col] = (_Float16)x[i];
  }
}

// ---------------------------------------------------------------------------
// WMMA GEMM:  C[M x N] = A[M x K] * Bt[N x K]^T (+ bias), f16 in, f32 acc.
// Block: 256 threads = 8 waves (4x2 wave grid), tile 128x128, BK=64.
// ---------------------------------------------------------------------------
#define GBM 128
#define GBN 128
#define GBK 64
#define GLD 72  // LDS row stride in halfs (144B, 16B multiple, conflict pad)

template <bool OUT_F16, bool HAS_BIAS>
__global__ void gemm_wmma_kernel(const _Float16* __restrict__ A, int lda,
                                 const _Float16* __restrict__ Bt, int ldb,
                                 void* __restrict__ Cv, int ldc,
                                 const float* __restrict__ bias, int M, int N, int K) {
  __shared__ _Float16 sA[GBM * GLD];
  __shared__ _Float16 sB[GBN * GLD];

  const int t = threadIdx.x;
  const int lane = t & 31, w = t >> 5;
  const int wm = w & 3, wn = w >> 2;  // 4 waves in M x 2 waves in N
  const int lr = lane & 15, hi4 = lane >> 4, kof = hi4 * 8;
  const int rowBlock = blockIdx.y * GBM;
  const int colBlock = blockIdx.x * GBN;

  v8f acc[2][4];
#pragma unroll
  for (int fm = 0; fm < 2; ++fm)
#pragma unroll
    for (int fn = 0; fn < 4; ++fn) acc[fm][fn] = v8f_zero();

  for (int k0 = 0; k0 < K; k0 += GBK) {
    // stage A/B tiles: 128 rows x 64 halfs = 1024 uint4 each; 4 per thread
#pragma unroll
    for (int j = 0; j < 4; ++j) {
      int lin = t + j * 256;  // 0..1023
      int r = lin >> 3;       // row 0..127
      int c = lin & 7;        // uint4 0..7
      *(uint4*)(&sA[r * GLD + c * 8]) =
          *(const uint4*)(A + (size_t)(rowBlock + r) * lda + k0 + c * 8);
      *(uint4*)(&sB[r * GLD + c * 8]) =
          *(const uint4*)(Bt + (size_t)(colBlock + r) * ldb + k0 + c * 8);
    }
    if (k0 + GBK < K) {  // CDNA5 global_prefetch_b8 of next K tile
      __builtin_prefetch(A + (size_t)(rowBlock + (t >> 1)) * lda + k0 + GBK + (t & 1) * 32, 0, 1);
      __builtin_prefetch(Bt + (size_t)(colBlock + (t >> 1)) * ldb + k0 + GBK + (t & 1) * 32, 0, 1);
    }
    __syncthreads();
#pragma unroll
    for (int ks = 0; ks < 2; ++ks) {
      int kb = ks * 32 + kof;
      v16h af[2], bfr[4];
#pragma unroll
      for (int fm = 0; fm < 2; ++fm)
        af[fm] = ldfrag(&sA[(wm * 32 + fm * 16 + lr) * GLD + kb]);
#pragma unroll
      for (int fn = 0; fn < 4; ++fn)
        bfr[fn] = ldfrag(&sB[(wn * 64 + fn * 16 + lr) * GLD + kb]);
#pragma unroll
      for (int fm = 0; fm < 2; ++fm)
#pragma unroll
        for (int fn = 0; fn < 4; ++fn)
          acc[fm][fn] = wmma16x16x32(af[fm], bfr[fn], acc[fm][fn]);
    }
    __syncthreads();
  }

  // store: C layout — VGPR e: lanes 0-15 -> m=e, lanes 16-31 -> m=e+8; n=lr
#pragma unroll
  for (int fm = 0; fm < 2; ++fm)
#pragma unroll
    for (int fn = 0; fn < 4; ++fn) {
      int col = colBlock + wn * 64 + fn * 16 + lr;
      int row0 = rowBlock + wm * 32 + fm * 16 + hi4 * 8;
      float bv = HAS_BIAS ? bias[col] : 0.f;
#pragma unroll
      for (int e = 0; e < 8; ++e) {
        float v = acc[fm][fn][e] + bv;
        if constexpr (OUT_F16)
          ((_Float16*)Cv)[(size_t)(row0 + e) * ldc + col] = (_Float16)v;
        else
          ((float*)Cv)[(size_t)(row0 + e) * ldc + col] = v;
      }
    }
}

// ---------------------------------------------------------------------------
// Attention: one block = 16 query rows x one (b,h). 128 threads = 4 waves,
// each wave owns a 16-key (phase 1) / 16-dh (phase 3) quarter.
// Dynamic LDS: S (16 x 1025 f32), P (16 x 1040 f16), red (256 f32) ~ 100 KB.
// ---------------------------------------------------------------------------
#define ASS 1025
#define APP 1040
#define ATTN_SMEM_BYTES (16 * ASS * 4 + 16 * APP * 2 + 256 * 4)

__global__ void attn_kernel(const _Float16* __restrict__ Q, const _Float16* __restrict__ Kf,
                            const _Float16* __restrict__ Vt, _Float16* __restrict__ O) {
  extern __shared__ char smem[];
  float* S = (float*)smem;
  _Float16* P = (_Float16*)(smem + 16 * ASS * 4);
  float* red = (float*)(smem + 16 * ASS * 4 + 16 * APP * 2);

  const int t = threadIdx.x;
  const int lane = t & 31, wn = t >> 5;  // wave owns key/dh quarter wn
  const int lr = lane & 15, hi4 = lane >> 4, kof = hi4 * 8;
  const int bh = blockIdx.y, b = bh >> 4, h = bh & 15;
  const int q0 = blockIdx.x * 16;

  // Q fragments (rows q0..q0+15, DH split into two 32-wide k-steps)
  const _Float16* qp = Q + ((size_t)(b * SEQ_N + q0 + lr)) * INNERv + h * DHEAD;
  v16h aq0 = ldfrag(qp + kof);
  v16h aq1 = ldfrag(qp + 32 + kof);

  // ---- Phase 1: S = (Q K^T) * scale, written to LDS ----
  for (int kt = 0; kt < 16; ++kt) {
    int kb0 = kt * 64 + wn * 16;  // 16 keys for this wave
    const _Float16* kp = Kf + ((size_t)(b * SEQ_M + kb0 + lr)) * INNERv + h * DHEAD;
    v16h bk0 = ldfrag(kp + kof);
    v16h bk1 = ldfrag(kp + 32 + kof);
    v8f s = v8f_zero();
    s = wmma16x16x32(aq0, bk0, s);
    s = wmma16x16x32(aq1, bk1, s);
#pragma unroll
    for (int e = 0; e < 8; ++e)
      S[(hi4 * 8 + e) * ASS + kb0 + lr] = s[e] * ATT_SCALE;
  }
  __syncthreads();

  // ---- Phase 2: softmax over 1024 keys, 8 segments of 128 per row ----
  {
    int row = t & 15, seg = t >> 4;  // 8 segs x 16 rows
    float* Srow = S + row * ASS + seg * 128;
    float mx = -3.0e38f;
    for (int i = 0; i < 128; ++i) mx = fmaxf(mx, Srow[i]);
    red[seg * 16 + row] = mx;
    __syncthreads();
    mx = red[row];
#pragma unroll
    for (int s2 = 1; s2 < 8; ++s2) mx = fmaxf(mx, red[s2 * 16 + row]);
    float sum = 0.f;
    for (int i = 0; i < 128; ++i) {
      float e = __expf(Srow[i] - mx);
      Srow[i] = e;
      sum += e;
    }
    red[128 + seg * 16 + row] = sum;
    __syncthreads();
    float tot = red[128 + row];
#pragma unroll
    for (int s2 = 1; s2 < 8; ++s2) tot += red[128 + s2 * 16 + row];
    float inv = 1.0f / tot;
    _Float16* Prow = P + row * APP + seg * 128;
    for (int i = 0; i < 128; ++i) Prow[i] = (_Float16)(Srow[i] * inv);
  }
  __syncthreads();

  // ---- Phase 3: O = P @ V  (contraction over 1024 keys, 32 k-steps) ----
  v8f o = v8f_zero();
  const _Float16* vbase = Vt + ((size_t)(b * INNERv + h * DHEAD + wn * 16 + lr)) * SEQ_M;
  for (int kk = 0; kk < 32; ++kk) {
    int kb = kk * 32 + kof;
    v16h ap = ldfrag(&P[lr * APP + kb]);
    v16h bv = ldfrag(vbase + kb);
    o = wmma16x16x32(ap, bv, o);
  }
#pragma unroll
  for (int e = 0; e < 8; ++e)
    O[((size_t)(b * SEQ_N + q0 + hi4 * 8 + e)) * INNERv + h * DHEAD + wn * 16 + lr] =
        (_Float16)o[e];
}

// ---------------------------------------------------------------------------
// Host-side launch
// ---------------------------------------------------------------------------
extern "C" void kernel_launch(void* const* d_in, const int* in_sizes, int n_in,
                              void* d_out, int out_size, void* d_ws, size_t ws_size,
                              hipStream_t stream) {
  (void)in_sizes; (void)n_in; (void)out_size; (void)ws_size;

  const float* person  = (const float*)d_in[0];
  const float* garment = (const float*)d_in[1];
  const float* Wq = (const float*)d_in[2];
  const float* Wk = (const float*)d_in[3];
  const float* Wv = (const float*)d_in[4];
  const float* Wo = (const float*)d_in[5];
  const float* bo = (const float*)d_in[6];
  const float* Wf = (const float*)d_in[7];
  const float* bf = (const float*)d_in[8];
  const float* gq = (const float*)d_in[9];
  const float* betaq = (const float*)d_in[10];
  const float* gk = (const float*)d_in[11];
  const float* betak = (const float*)d_in[12];
  float* out = (float*)d_out;

  _Float16* w0 = (_Float16*)d_ws;
  size_t off = 0;
  _Float16* pLN = w0 + off; off += (size_t)BATCH * SEQ_N * DQv;
  _Float16* gLN = w0 + off; off += (size_t)BATCH * SEQ_M * DCv;
  _Float16* WqT = w0 + off; off += (size_t)INNERv * DQv;
  _Float16* WkT = w0 + off; off += (size_t)INNERv * DCv;
  _Float16* WvT = w0 + off; off += (size_t)INNERv * DCv;
  _Float16* WoT = w0 + off; off += (size_t)DQv * INNERv;
  _Float16* WfT = w0 + off; off += (size_t)DQv * (2 * DQv);
  _Float16* Qh  = w0 + off; off += (size_t)BATCH * SEQ_N * INNERv;
  _Float16* Kh  = w0 + off; off += (size_t)BATCH * SEQ_M * INNERv;
  _Float16* Vh  = w0 + off; off += (size_t)BATCH * SEQ_M * INNERv;
  _Float16* VtB = w0 + off; off += (size_t)BATCH * INNERv * SEQ_M;
  _Float16* Att = w0 + off; off += (size_t)BATCH * SEQ_N * INNERv;
  _Float16* Cat = w0 + off; off += (size_t)BATCH * SEQ_N * 2 * DQv;

  dim3 tb32(32, 8);

  // Weight transpose+convert
  wtrans_kernel<<<dim3(INNERv / 32, DQv / 32), tb32, 0, stream>>>(Wq, WqT, DQv, INNERv);
  wtrans_kernel<<<dim3(INNERv / 32, DCv / 32), tb32, 0, stream>>>(Wk, WkT, DCv, INNERv);
  wtrans_kernel<<<dim3(INNERv / 32, DCv / 32), tb32, 0, stream>>>(Wv, WvT, DCv, INNERv);
  wtrans_kernel<<<dim3(DQv / 32, INNERv / 32), tb32, 0, stream>>>(Wo, WoT, INNERv, DQv);
  wtrans_kernel<<<dim3(DQv / 32, (2 * DQv) / 32), tb32, 0, stream>>>(Wf, WfT, 2 * DQv, DQv);

  // LayerNorms
  ln_f16_kernel<DQv><<<BATCH * SEQ_N, 256, 0, stream>>>(person, gq, betaq, pLN);
  ln_f16_kernel<DCv><<<BATCH * SEQ_M, 256, 0, stream>>>(garment, gk, betak, gLN);

  // Q/K/V projections (f16 out)
  gemm_wmma_kernel<true, false><<<dim3(INNERv / GBN, (BATCH * SEQ_N) / GBM), 256, 0, stream>>>(
      pLN, DQv, WqT, DQv, (void*)Qh, INNERv, nullptr, BATCH * SEQ_N, INNERv, DQv);
  gemm_wmma_kernel<true, false><<<dim3(INNERv / GBN, (BATCH * SEQ_M) / GBM), 256, 0, stream>>>(
      gLN, DCv, WkT, DCv, (void*)Kh, INNERv, nullptr, BATCH * SEQ_M, INNERv, DCv);
  gemm_wmma_kernel<true, false><<<dim3(INNERv / GBN, (BATCH * SEQ_M) / GBM), 256, 0, stream>>>(
      gLN, DCv, WvT, DCv, (void*)Vh, INNERv, nullptr, BATCH * SEQ_M, INNERv, DCv);

  // V transpose for contiguous PV fragments
  vtrans_kernel<<<dim3(INNERv / 32, SEQ_M / 32, BATCH), tb32, 0, stream>>>(Vh, VtB);

  // Residual into concat cols [0, DQ)
  {
    size_t total = (size_t)BATCH * SEQ_N * DQv;
    cvt_resid_kernel<<<(unsigned)((total + 255) / 256), 256, 0, stream>>>(person, Cat);
  }

  // Attention (flash-style, full score row in LDS)
  attn_kernel<<<dim3(SEQ_N / 16, BATCH * NHEAD), 128, ATTN_SMEM_BYTES, stream>>>(
      Qh, Kh, VtB, Att);

  // att @ Wo + bo -> concat cols [DQ, 2*DQ) (f16)
  gemm_wmma_kernel<true, true><<<dim3(DQv / GBN, (BATCH * SEQ_N) / GBM), 256, 0, stream>>>(
      Att, INNERv, WoT, INNERv, (void*)(Cat + DQv), 2 * DQv, bo, BATCH * SEQ_N, DQv, INNERv);

  // concat @ Wf + bf -> out (f32)
  gemm_wmma_kernel<false, true><<<dim3(DQv / GBN, (BATCH * SEQ_N) / GBM), 256, 0, stream>>>(
      Cat, 2 * DQv, WfT, 2 * DQv, (void*)out, DQv, bf, BATCH * SEQ_N, DQv, 2 * DQv);
}
